// SelfAttention_79602923864662
// MI455X (gfx1250) — compile-verified, matrix operand order checked by reading, not run on hardware
//
#include <hip/hip_runtime.h>

typedef __attribute__((ext_vector_type(16))) _Float16 v16h;
typedef __attribute__((ext_vector_type(8)))  float    v8f;

#define S_LEN 4096
#define C_CH  256

union H8 { uint4 u; _Float16 h[8]; };

// CDNA5 async DMA: global -> LDS, 16B per lane, tracked by ASYNCcnt.
// VDST operand = LDS byte address (low 32 bits of the generic pointer).
__device__ __forceinline__ void async_copy_b128(void* lds_dst, const void* gsrc) {
  unsigned off = (unsigned)(uintptr_t)lds_dst;
  asm volatile("global_load_async_to_lds_b128 %0, %1, off"
               :: "v"(off), "v"(gsrc) : "memory");
}
__device__ __forceinline__ void wait_async() {
  asm volatile("s_wait_asynccnt 0x0" ::: "memory");
}

// Load a 16x32 (rows x K) f16 fragment for WMMA from LDS, given a pointer to
// this lane's row. Matches CDNA5 ISA 16-bit A/B layout:
//   lanes 0-15 : VGPR0-3 hold K=0..7 pairs,  VGPR4-7 hold K=16..23 pairs
//   lanes 16-31: VGPR0-3 hold K=8..15 pairs, VGPR4-7 hold K=24..31 pairs
__device__ __forceinline__ v16h frag_k32(const _Float16* rowPtr) {
  const int lane = threadIdx.x & 31;
  const int kb = (lane >> 4) << 3;
  v16h f;
#pragma unroll
  for (int i = 0; i < 8; ++i) {
    const int k = ((i < 4) ? (2 * i) : (16 + 2 * (i - 4))) + kb;
    f[2 * i]     = rowPtr[k];
    f[2 * i + 1] = rowPtr[k + 1];
  }
  return f;
}

// ---------------- weight conversion f32 -> f16 ----------------
__global__ __launch_bounds__(256) void cvt_kernel(const float* __restrict__ wq,
                                                  const float* __restrict__ wo,
                                                  _Float16* __restrict__ wqh,
                                                  _Float16* __restrict__ woh) {
  int i = blockIdx.x * 256 + threadIdx.x;
  if (i < 768 * 256) wqh[i] = (_Float16)wq[i];
  if (i < 256 * 256) woh[i] = (_Float16)wo[i];
}

// ---------------- GroupNorm (32 groups of 8 channels) ----------------
__global__ __launch_bounds__(256) void gn_kernel(const float* __restrict__ x,
                                                 const float* __restrict__ gamma,
                                                 const float* __restrict__ beta,
                                                 _Float16* __restrict__ normed) {
  __shared__ float ssum[256];
  __shared__ float ssq[256];
  const int bg = blockIdx.x;
  const int b = bg >> 5, g = bg & 31;
  const size_t base = ((size_t)b * C_CH + g * 8) * S_LEN;
  const int tid = threadIdx.x;
  float s = 0.f, q = 0.f;
#pragma unroll
  for (int it = 0; it < 32; ++it) {           // 8*4096/256 iterations, vector f32x4
    const int i = (tid + it * 256) * 4;
    const float4 v = *(const float4*)(x + base + i);
    s += v.x + v.y + v.z + v.w;
    q += v.x * v.x + v.y * v.y + v.z * v.z + v.w * v.w;
  }
  ssum[tid] = s;
  ssq[tid] = q;
  __syncthreads();
  for (int off = 128; off > 0; off >>= 1) {
    if (tid < off) {
      ssum[tid] += ssum[tid + off];
      ssq[tid] += ssq[tid + off];
    }
    __syncthreads();
  }
  const float n = 8.f * (float)S_LEN;
  const float mean = ssum[0] / n;
  const float var = ssq[0] / n - mean * mean;
  const float inv = rsqrtf(var + 1e-5f);
#pragma unroll
  for (int it = 0; it < 32; ++it) {
    const int i = (tid + it * 256) * 4;
    const int c = g * 8 + (i >> 12);
    const float gsc = inv * gamma[c];
    const float bof = beta[c] - mean * gsc;
    const float4 v = *(const float4*)(x + base + i);
    H8 o;  // reuse as 4-half pack via first half of union
    o.h[0] = (_Float16)(v.x * gsc + bof);
    o.h[1] = (_Float16)(v.y * gsc + bof);
    o.h[2] = (_Float16)(v.z * gsc + bof);
    o.h[3] = (_Float16)(v.w * gsc + bof);
    *(uint2*)(normed + base + i) = make_uint2(o.u.x, o.u.y);
  }
}

// ---------------- QKV projection GEMM: Y[b,m,n] = sum_k W[m,k] * X[b,k,n] ----------------
// M=768, K=256, N=4096. Block: 64x64 tile, 4 waves, wave w -> rows [16w,16w+16).
__global__ __launch_bounds__(128) void qkv_gemm(const _Float16* __restrict__ W,
                                                const _Float16* __restrict__ X,
                                                _Float16* __restrict__ Y) {
  __shared__ _Float16 As[64][40];  // [m][k]
  __shared__ _Float16 Bs[64][40];  // [n][k]  (transposed so K-pairs are contiguous)
  const int n0 = blockIdx.x * 64;
  const int m0 = blockIdx.y * 64;
  const int b = blockIdx.z;
  const int tid = threadIdx.x;
  const int wave = tid >> 5, lane = tid & 31;
  const int r15 = lane & 15, hh = lane >> 4;
  v8f acc[4] = {};
  for (int k0 = 0; k0 < 256; k0 += 32) {
    __syncthreads();
    // A tile: direct copy -> async DMA (64 rows x 32 halfs, 16B chunks)
#pragma unroll
    for (int it = 0; it < 2; ++it) {
      const int c = tid + it * 128;              // 0..255
      const int m = c >> 2, qv = (c & 3) * 8;
      async_copy_b128(&As[m][qv], W + (size_t)(m0 + m) * 256 + k0 + qv);
    }
    // B tile: transposed staging, vectorized global b128 reads
#pragma unroll
    for (int it = 0; it < 2; ++it) {
      const int c = tid + it * 128;              // 0..255
      const int k = c >> 3, nq = (c & 7) * 8;
      H8 v;
      v.u = *(const uint4*)(X + ((size_t)b * C_CH + k0 + k) * S_LEN + n0 + nq);
#pragma unroll
      for (int e = 0; e < 8; ++e) Bs[nq + e][k] = v.h[e];
    }
    wait_async();
    __syncthreads();
    const v16h a = frag_k32(&As[wave * 16 + r15][0]);
#pragma unroll
    for (int j = 0; j < 4; ++j) {
      const v16h bb = frag_k32(&Bs[j * 16 + r15][0]);
      acc[j] = __builtin_amdgcn_wmma_f32_16x16x32_f16(false, a, false, bb,
                                                      (short)0, acc[j], false, false);
    }
  }
#pragma unroll
  for (int j = 0; j < 4; ++j) {
    const int n = n0 + j * 16 + r15;
#pragma unroll
    for (int r = 0; r < 8; ++r) {
      const int m = m0 + wave * 16 + r + 8 * hh;
      Y[((size_t)b * 768 + m) * S_LEN + n] = (_Float16)acc[j][r];
    }
  }
}

// ---------------- Flash attention ----------------
// grid: (S/64, B*N_HEAD). Block 128 = 4 waves; wave w owns 16 query rows.
// qkv layout: [b, head*192 + {0..63:q, 64..127:k, 128..191:v}, s]
__global__ __launch_bounds__(128) void attn_kernel(const _Float16* __restrict__ qkv,
                                                   _Float16* __restrict__ attnout) {
  __shared__ _Float16 Qs[64][72];       // [query][d]
  __shared__ _Float16 KsT[32][72];      // [key][d]
  __shared__ _Float16 Vs[64][40];       // [d][key]
  __shared__ _Float16 Ps[4][16][40];    // per-wave probs [query][key]
  const int s0 = blockIdx.x * 64;
  const int bh = blockIdx.y;
  const int b = bh >> 2, h = bh & 3;
  const _Float16* qb  = qkv + ((size_t)b * 768 + h * 192) * S_LEN;
  const _Float16* kb_ = qb + (size_t)64 * S_LEN;
  const _Float16* vb  = qb + (size_t)128 * S_LEN;
  const int tid = threadIdx.x;
  const int wave = tid >> 5, lane = tid & 31;
  const int r15 = lane & 15, hh = lane >> 4;

  // stage Q tile (transposed: [query][d]); vectorized b128 reads
#pragma unroll
  for (int it = 0; it < 4; ++it) {
    const int c = tid + it * 128;                // 0..511
    const int d = c >> 3, jq = (c & 7) * 8;
    H8 v;
    v.u = *(const uint4*)(qb + (size_t)d * S_LEN + s0 + jq);
#pragma unroll
    for (int e = 0; e < 8; ++e) Qs[jq + e][d] = v.h[e];
  }
  __syncthreads();
  const v16h aQ0 = frag_k32(&Qs[wave * 16 + r15][0]);    // d 0..31
  const v16h aQ1 = frag_k32(&Qs[wave * 16 + r15][32]);   // d 32..63

  v8f o0 = {}, o1 = {}, o2 = {}, o3 = {};
  float rowm[8], rowl[8];
#pragma unroll
  for (int r = 0; r < 8; ++r) { rowm[r] = -1e30f; rowl[r] = 0.f; }

  for (int kt = 0; kt < 128; ++kt) {
    const int t0 = kt * 32;
    __syncthreads();
    // K tile: transposed staging (vectorized); V tile: async DMA (no transpose)
#pragma unroll
    for (int it = 0; it < 2; ++it) {
      const int c = tid + it * 128;              // 0..255
      const int d = c >> 2, qv = (c & 3) * 8;
      H8 v;
      v.u = *(const uint4*)(kb_ + (size_t)d * S_LEN + t0 + qv);
#pragma unroll
      for (int e = 0; e < 8; ++e) KsT[qv + e][d] = v.h[e];
      async_copy_b128(&Vs[d][qv], vb + (size_t)d * S_LEN + t0 + qv);
    }
    // prefetch next key/value tile rows into cache
    if (t0 + 32 < S_LEN) {
      const int d = tid & 63;
      const _Float16* p = (tid < 64) ? &kb_[(size_t)d * S_LEN + t0 + 32]
                                     : &vb[(size_t)d * S_LEN + t0 + 32];
      __builtin_prefetch(p, 0, 1);
    }
    wait_async();
    __syncthreads();

    // scores = Q (16x64) . K^T (64x32), scaled by 1/sqrt(C)=1/16
    const v16h bK0a = frag_k32(&KsT[r15][0]);
    const v16h bK0b = frag_k32(&KsT[r15][32]);
    const v16h bK1a = frag_k32(&KsT[16 + r15][0]);
    const v16h bK1b = frag_k32(&KsT[16 + r15][32]);
    v8f sc0 = {}, sc1 = {};
    sc0 = __builtin_amdgcn_wmma_f32_16x16x32_f16(false, aQ0, false, bK0a, (short)0, sc0, false, false);
    sc0 = __builtin_amdgcn_wmma_f32_16x16x32_f16(false, aQ1, false, bK0b, (short)0, sc0, false, false);
    sc1 = __builtin_amdgcn_wmma_f32_16x16x32_f16(false, aQ0, false, bK1a, (short)0, sc1, false, false);
    sc1 = __builtin_amdgcn_wmma_f32_16x16x32_f16(false, aQ1, false, bK1b, (short)0, sc1, false, false);

    // online softmax: D layout row m = r + 8*hh, column n = r15 (+16 for sc1)
#pragma unroll
    for (int r = 0; r < 8; ++r) {
      float a0 = sc0[r] * 0.0625f;
      float a1 = sc1[r] * 0.0625f;
      float mx = fmaxf(a0, a1);
#pragma unroll
      for (int off = 1; off < 16; off <<= 1) mx = fmaxf(mx, __shfl_xor(mx, off, 32));
      const float mnew = fmaxf(rowm[r], mx);
      const float corr = __expf(rowm[r] - mnew);
      rowm[r] = mnew;
      const float p0 = __expf(a0 - mnew);
      const float p1 = __expf(a1 - mnew);
      float ps = p0 + p1;
#pragma unroll
      for (int off = 1; off < 16; off <<= 1) ps += __shfl_xor(ps, off, 32);
      rowl[r] = rowl[r] * corr + ps;
      o0[r] *= corr; o1[r] *= corr; o2[r] *= corr; o3[r] *= corr;
      const int m = r + 8 * hh;
      Ps[wave][m][r15]      = (_Float16)p0;
      Ps[wave][m][r15 + 16] = (_Float16)p1;
    }
    __syncthreads();

    // O += P (16x32) . V^T (32x64)
    const v16h aP = frag_k32(&Ps[wave][r15][0]);
    const v16h bV0 = frag_k32(&Vs[r15][0]);
    const v16h bV1 = frag_k32(&Vs[16 + r15][0]);
    const v16h bV2 = frag_k32(&Vs[32 + r15][0]);
    const v16h bV3 = frag_k32(&Vs[48 + r15][0]);
    o0 = __builtin_amdgcn_wmma_f32_16x16x32_f16(false, aP, false, bV0, (short)0, o0, false, false);
    o1 = __builtin_amdgcn_wmma_f32_16x16x32_f16(false, aP, false, bV1, (short)0, o1, false, false);
    o2 = __builtin_amdgcn_wmma_f32_16x16x32_f16(false, aP, false, bV2, (short)0, o2, false, false);
    o3 = __builtin_amdgcn_wmma_f32_16x16x32_f16(false, aP, false, bV3, (short)0, o3, false, false);
  }

  // write O / rowsum ; attnout layout [b, head*64 + d, s] (f16)
  // the 8 rows per lane are contiguous in s -> one b128 store per d-tile
#pragma unroll
  for (int j = 0; j < 4; ++j) {
    const v8f oj = (j == 0) ? o0 : (j == 1) ? o1 : (j == 2) ? o2 : o3;
    const int d = j * 16 + r15;
    H8 v;
#pragma unroll
    for (int r = 0; r < 8; ++r) v.h[r] = (_Float16)(oj[r] / rowl[r]);
    *(uint4*)(attnout + ((size_t)b * C_CH + h * 64 + d) * S_LEN +
              s0 + wave * 16 + 8 * hh) = v.u;
  }
}

// ---------------- Output projection + bias + residual (f32 out) ----------------
__global__ __launch_bounds__(128) void out_gemm(const _Float16* __restrict__ W,
                                                const _Float16* __restrict__ X,
                                                const float* __restrict__ bias,
                                                const float* __restrict__ resid,
                                                float* __restrict__ Y) {
  __shared__ _Float16 As[64][40];
  __shared__ _Float16 Bs[64][40];
  const int n0 = blockIdx.x * 64;
  const int m0 = blockIdx.y * 64;
  const int b = blockIdx.z;
  const int tid = threadIdx.x;
  const int wave = tid >> 5, lane = tid & 31;
  const int r15 = lane & 15, hh = lane >> 4;
  v8f acc[4] = {};
  for (int k0 = 0; k0 < 256; k0 += 32) {
    __syncthreads();
#pragma unroll
    for (int it = 0; it < 2; ++it) {
      const int c = tid + it * 128;
      const int m = c >> 2, qv = (c & 3) * 8;
      async_copy_b128(&As[m][qv], W + (size_t)(m0 + m) * 256 + k0 + qv);
    }
#pragma unroll
    for (int it = 0; it < 2; ++it) {
      const int c = tid + it * 128;
      const int k = c >> 3, nq = (c & 7) * 8;
      H8 v;
      v.u = *(const uint4*)(X + ((size_t)b * C_CH + k0 + k) * S_LEN + n0 + nq);
#pragma unroll
      for (int e = 0; e < 8; ++e) Bs[nq + e][k] = v.h[e];
    }
    wait_async();
    __syncthreads();
    const v16h a = frag_k32(&As[wave * 16 + r15][0]);
#pragma unroll
    for (int j = 0; j < 4; ++j) {
      const v16h bb = frag_k32(&Bs[j * 16 + r15][0]);
      acc[j] = __builtin_amdgcn_wmma_f32_16x16x32_f16(false, a, false, bb,
                                                      (short)0, acc[j], false, false);
    }
  }
#pragma unroll
  for (int j = 0; j < 4; ++j) {
    const int n = n0 + j * 16 + r15;
#pragma unroll
    for (int r = 0; r < 8; ++r) {
      const int m = m0 + wave * 16 + r + 8 * hh;
      const size_t idx = ((size_t)b * C_CH + m) * S_LEN + n;
      Y[idx] = acc[j][r] + bias[m] + resid[idx];
    }
  }
}

extern "C" void kernel_launch(void* const* d_in, const int* in_sizes, int n_in,
                              void* d_out, int out_size, void* d_ws, size_t ws_size,
                              hipStream_t stream) {
  const float* x     = (const float*)d_in[0];
  const float* gamma = (const float*)d_in[1];
  const float* beta  = (const float*)d_in[2];
  const float* w_qkv = (const float*)d_in[3];
  const float* w_out = (const float*)d_in[4];
  const float* b_out = (const float*)d_in[5];
  float* out = (float*)d_out;

  char* ws = (char*)d_ws;
  _Float16* normed_h  = (_Float16*)(ws);                         //  4,194,304 B
  _Float16* qkv_h     = (_Float16*)(ws + 4194304);               // 12,582,912 B
  _Float16* attnout_h = (_Float16*)(ws + 4194304 + 12582912);    //  4,194,304 B
  _Float16* wqkv_h    = (_Float16*)(ws + 20971520);              //    393,216 B
  _Float16* wout_h    = (_Float16*)(ws + 20971520 + 393216);     //    131,072 B

  cvt_kernel<<<768, 256, 0, stream>>>(w_qkv, w_out, wqkv_h, wout_h);
  gn_kernel<<<64, 256, 0, stream>>>(x, gamma, beta, normed_h);
  qkv_gemm<<<dim3(64, 12, 2), 128, 0, stream>>>(wqkv_h, normed_h, qkv_h);
  attn_kernel<<<dim3(64, 8), 128, 0, stream>>>(qkv_h, attnout_h);
  out_gemm<<<dim3(64, 4, 2), 128, 0, stream>>>(wout_h, attnout_h, b_out, x, out);
}
